// MyModel_66408784330987
// MI455X (gfx1250) — compile-verified, matrix operand order checked by reading, not run on hardware
//
#include <hip/hip_runtime.h>
#include <math.h>

typedef __bf16 bf16;
typedef __attribute__((ext_vector_type(16))) __bf16 v16bf;
typedef __attribute__((ext_vector_type(8)))  float  v8f;

#define NSEQ 4096
#define DMODEL 768
#define FDIM 3072
#define NHEAD 12
#define DHEAD 64

#if defined(__has_builtin)
#if __has_builtin(__builtin_amdgcn_s_wait_asynccnt)
#define USE_ASYNC 1
#endif
#endif
#ifndef USE_ASYNC
#define USE_ASYNC 0
#endif

struct Tile16 { union { v16bf v; uint4 u[2]; }; };

// Load a 16x32 bf16 tile in the WMMA A/B register layout (ISA 7.12.2):
// lane r = L&15 holds row (row+r); khi = L>>4 selects the 8-wide K half.
__device__ __forceinline__ v16bf load_tile(const bf16* __restrict__ base, int row,
                                           int ld, int kb, int lane) {
  int r = lane & 15, khi = lane >> 4;
  const bf16* p = base + (size_t)(row + r) * (size_t)ld + kb + 8 * khi;
  Tile16 t;
  t.u[0] = *(const uint4*)(p);
  t.u[1] = *(const uint4*)(p + 16);
  return t.v;
}

// Same layout but from an LDS-resident 16-row x 32-col tile array (ld fixed 32).
__device__ __forceinline__ v16bf load_tile_lds(const bf16* base, int row, int lane) {
  int r = lane & 15, khi = lane >> 4;
  const bf16* p = base + (row + r) * 32 + 8 * khi;
  Tile16 t;
  t.u[0] = *(const uint4*)(p);
  t.u[1] = *(const uint4*)(p + 16);
  return t.v;
}

__device__ __forceinline__ v8f wmma_bf16(v16bf a, v16bf b, v8f c) {
  return __builtin_amdgcn_wmma_f32_16x16x32_bf16(false, a, false, b, (short)0, c,
                                                 false, false);
}

// Async copy of one 16-byte chunk global -> LDS (Tensor-less async path,
// tracked by ASYNCcnt). vdst = LDS byte address (low 32 bits of the generic
// pointer, per the flat-aperture rule), vaddr = 64-bit global address.
__device__ __forceinline__ void async_b128(const bf16* gptr, bf16* lptr) {
  asm volatile("global_load_async_to_lds_b128 %0, %1, off"
               :: "v"((unsigned)(size_t)lptr), "v"(gptr)
               : "memory");
}

// ---------------- weight conversion kernels (f32 -> bf16, transposed) ----------

// Wq/Wk/Wv: [H, D, DH] f32 -> Wt [H*DH, D] bf16  (Wt[c, d] = W[c/64, d, c%64])
__global__ void conv_qkv_w(const float* __restrict__ W, bf16* __restrict__ Wt) {
  size_t idx = (size_t)blockIdx.x * 256 + threadIdx.x;  // DMODEL*DMODEL total
  int c = (int)(idx / DMODEL), d = (int)(idx % DMODEL);
  Wt[idx] = (bf16)W[(size_t)(c >> 6) * (DMODEL * DHEAD) + (size_t)d * DHEAD + (c & 63)];
}

// W [R, C] f32 -> Wt [C, R] bf16
__global__ void transpose_w(const float* __restrict__ W, bf16* __restrict__ Wt,
                            int R, int C) {
  size_t idx = (size_t)blockIdx.x * 256 + threadIdx.x;  // R*C total
  int c = (int)(idx / R), r = (int)(idx % R);
  Wt[idx] = (bf16)W[(size_t)r * C + c];
}

// ---------------- LayerNorm: f32 [N, 768] -> bf16 [N, 768] --------------------
__global__ void ln_kernel(const float* __restrict__ x, const float* __restrict__ g,
                          const float* __restrict__ b, bf16* __restrict__ out) {
  __shared__ float s1[256], s2[256];
  int row = blockIdx.x, t = threadIdx.x;
  const float* xr = x + (size_t)row * DMODEL;
  float v0 = xr[t], v1 = xr[t + 256], v2 = xr[t + 512];
  s1[t] = v0 + v1 + v2;
  s2[t] = v0 * v0 + v1 * v1 + v2 * v2;
  __syncthreads();
  for (int o = 128; o > 0; o >>= 1) {
    if (t < o) { s1[t] += s1[t + o]; s2[t] += s2[t + o]; }
    __syncthreads();
  }
  float mean = s1[0] * (1.0f / DMODEL);
  float var  = s2[0] * (1.0f / DMODEL) - mean * mean;
  float rstd = rsqrtf(var + 1e-5f);
  bf16* orow = out + (size_t)row * DMODEL;
  orow[t]       = (bf16)((v0 - mean) * rstd * g[t]       + b[t]);
  orow[t + 256] = (bf16)((v1 - mean) * rstd * g[t + 256] + b[t + 256]);
  orow[t + 512] = (bf16)((v2 - mean) * rstd * g[t + 512] + b[t + 512]);
}

// ---------------- generic WMMA GEMM: out = epilogue(A @ Bt^T + bias) ----------
// A: bf16 [M, Kd] row-major; Bt: bf16 [Ncol, Kd] row-major (i.e. B transposed).
// Block: 128 threads (4 waves), computes 128 rows x 64 cols of C.
// Each wave: 32 rows x 64 cols (2 M-tiles x 4 N-tiles, 8 WMMA accums).
// B tiles (64 rows x 32 k) are staged in LDS, double-buffered, via
// global_load_async_to_lds_b128 (ASYNCcnt + s_wait_asynccnt).
// EPI 0: bf16 out [M, Ncol], +bias
// EPI 1: bf16 out [Ncol, M] (transposed store), +bias
// EPI 2: f32  out [M, Ncol], +bias +res
// EPI 3: bf16 out [M, Ncol], gelu(x+bias) exact
template <int EPI>
__global__ void __launch_bounds__(128)
gemm_wmma(const bf16* __restrict__ A, const bf16* __restrict__ Bt,
          const float* __restrict__ bias, const float* __restrict__ res,
          void* __restrict__ outp, int M, int Kd, int Ncol) {
  __shared__ bf16 bsh[2][64 * 32];  // two 4KB B-tile buffers
  int tid = threadIdx.x;
  int lane = tid & 31, wave = tid >> 5;
  int mr = blockIdx.x * 128 + wave * 32;
  int cb = blockIdx.y * 64;

  // Each thread stages two 16B chunks of the 64x32 B tile.
  // chunk ch in [0,256): row = ch>>2, 8-elem segment = ch&3.
  int ch0 = tid * 2, ch1 = tid * 2 + 1;
  int r0 = ch0 >> 2, sg0 = ch0 & 3;
  int r1 = ch1 >> 2, sg1 = ch1 & 3;

  auto stage = [&](int buf, int kb) {
    const bf16* g0 = Bt + (size_t)(cb + r0) * Kd + kb + sg0 * 8;
    const bf16* g1 = Bt + (size_t)(cb + r1) * Kd + kb + sg1 * 8;
    bf16* l0 = &bsh[buf][r0 * 32 + sg0 * 8];
    bf16* l1 = &bsh[buf][r1 * 32 + sg1 * 8];
#if USE_ASYNC
    async_b128(g0, l0);
    async_b128(g1, l1);
#else
    *(uint4*)l0 = *(const uint4*)g0;
    *(uint4*)l1 = *(const uint4*)g1;
#endif
  };

  v8f acc[2][4];
#pragma unroll
  for (int mt = 0; mt < 2; ++mt)
#pragma unroll
    for (int t = 0; t < 4; ++t)
#pragma unroll
      for (int i = 0; i < 8; ++i) acc[mt][t][i] = 0.0f;

  int nk = Kd / 32;
  stage(0, 0);
  for (int i = 0; i < nk; ++i) {
    int kb = i * 32;
    int cur = i & 1;
    if (i + 1 < nk) {
      stage(cur ^ 1, kb + 32);  // prefetch next k-step into other buffer
#if USE_ASYNC
      __builtin_amdgcn_s_wait_asynccnt(2);  // cur buffer's 2 issues retired
#endif
    } else {
#if USE_ASYNC
      __builtin_amdgcn_s_wait_asynccnt(0);
#endif
    }
    __syncthreads();  // whole B tile visible to all waves

    v16bf a0 = load_tile(A, mr, Kd, kb, lane);
    v16bf a1 = load_tile(A, mr + 16, Kd, kb, lane);
#pragma unroll
    for (int t = 0; t < 4; ++t) {
      v16bf b = load_tile_lds(&bsh[cur][0], 16 * t, lane);
      acc[0][t] = wmma_bf16(a0, b, acc[0][t]);
      acc[1][t] = wmma_bf16(a1, b, acc[1][t]);
    }
    __syncthreads();  // all reads done before buffer is overwritten
  }

  int n = lane & 15, hi = lane >> 4;
#pragma unroll
  for (int mt = 0; mt < 2; ++mt)
#pragma unroll
    for (int t = 0; t < 4; ++t) {
      int col = cb + 16 * t + n;
      float bc = bias[col];
#pragma unroll
      for (int v = 0; v < 8; ++v) {
        int row = mr + mt * 16 + v + 8 * hi;
        float val = acc[mt][t][v] + bc;
        if (EPI == 0) {
          ((bf16*)outp)[(size_t)row * Ncol + col] = (bf16)val;
        } else if (EPI == 1) {
          ((bf16*)outp)[(size_t)col * M + row] = (bf16)val;
        } else if (EPI == 2) {
          ((float*)outp)[(size_t)row * Ncol + col] =
              val + res[(size_t)row * Ncol + col];
        } else {
          float gg = 0.5f * val * (1.0f + erff(val * 0.70710678118654752f));
          ((bf16*)outp)[(size_t)row * Ncol + col] = (bf16)gg;
        }
      }
    }
}

// ---------------- flash attention ---------------------------------------------
// Q, K: bf16 [N, H*64] row-major; Vt: bf16 [H*64, N] row-major.
// Each wave: 16 query rows x DH=64 for one head; stream keys 32 at a time.
__global__ void __launch_bounds__(128)
attn_kernel(const bf16* __restrict__ Q, const bf16* __restrict__ Kb,
            const bf16* __restrict__ Vt, bf16* __restrict__ O) {
  __shared__ bf16 plds[4][16 * 32];
  int lane = threadIdx.x & 31, wave = threadIdx.x >> 5;
  int h = blockIdx.y;
  int n0 = blockIdx.x * 64 + wave * 16;
  const bf16* qbase = Q + h * DHEAD;
  const bf16* kbase = Kb + h * DHEAD;
  const bf16* vbase = Vt + (size_t)(h * DHEAD) * NSEQ;

  v16bf qa0 = load_tile(qbase, n0, DMODEL, 0, lane);
  v16bf qa1 = load_tile(qbase, n0, DMODEL, 32, lane);

  float Mv[8], Ls[8];
  v8f acc[4];
#pragma unroll
  for (int v = 0; v < 8; ++v) { Mv[v] = -3.0e38f; Ls[v] = 0.0f; }
#pragma unroll
  for (int t = 0; t < 4; ++t)
#pragma unroll
    for (int i = 0; i < 8; ++i) acc[t][i] = 0.0f;

  int n = lane & 15, hi = lane >> 4;
  bf16* myl = plds[wave];

  for (int j = 0; j < NSEQ; j += 32) {
    // scores: S[16 x 32] = Q[16 x 64] @ K[j..j+31, 64]^T
    v16bf k00 = load_tile(kbase, j,      DMODEL, 0,  lane);
    v16bf k01 = load_tile(kbase, j,      DMODEL, 32, lane);
    v16bf k10 = load_tile(kbase, j + 16, DMODEL, 0,  lane);
    v16bf k11 = load_tile(kbase, j + 16, DMODEL, 32, lane);
    v8f s0, s1;
#pragma unroll
    for (int i = 0; i < 8; ++i) { s0[i] = 0.0f; s1[i] = 0.0f; }
    s0 = wmma_bf16(qa0, k00, s0);
    s0 = wmma_bf16(qa1, k01, s0);
    s1 = wmma_bf16(qa0, k10, s1);
    s1 = wmma_bf16(qa1, k11, s1);

    // online softmax: rows live in one VGPR across each 16-lane half
    float p0[8], p1[8], sc[8];
#pragma unroll
    for (int v = 0; v < 8; ++v) {
      float a0 = s0[v] * 0.125f, a1 = s1[v] * 0.125f;  // 1/sqrt(64)
      float rm = fmaxf(a0, a1);
      rm = fmaxf(rm, __shfl_xor(rm, 1));
      rm = fmaxf(rm, __shfl_xor(rm, 2));
      rm = fmaxf(rm, __shfl_xor(rm, 4));
      rm = fmaxf(rm, __shfl_xor(rm, 8));
      float nm = fmaxf(Mv[v], rm);
      float scale = __expf(Mv[v] - nm);
      float e0 = __expf(a0 - nm), e1 = __expf(a1 - nm);
      float rs = e0 + e1;
      rs += __shfl_xor(rs, 1);
      rs += __shfl_xor(rs, 2);
      rs += __shfl_xor(rs, 4);
      rs += __shfl_xor(rs, 8);
      Ls[v] = Ls[v] * scale + rs;
      Mv[v] = nm;
      p0[v] = e0; p1[v] = e1; sc[v] = scale;
    }
#pragma unroll
    for (int t = 0; t < 4; ++t)
#pragma unroll
      for (int v = 0; v < 8; ++v) acc[t][v] *= sc[v];

    // re-layout P (C layout -> A layout) through LDS; DS ops are in-order per wave
#pragma unroll
    for (int v = 0; v < 8; ++v) {
      int r = v + 8 * hi;
      myl[r * 32 + n]      = (bf16)p0[v];
      myl[r * 32 + 16 + n] = (bf16)p1[v];
    }
    int rr = lane & 15, khi = lane >> 4;
    Tile16 pt;
    pt.u[0] = *(const uint4*)&myl[rr * 32 + 8 * khi];
    pt.u[1] = *(const uint4*)&myl[rr * 32 + 16 + 8 * khi];

    // O += P[16 x 32] @ V[j..j+31, 64]
#pragma unroll
    for (int t = 0; t < 4; ++t) {
      v16bf bv = load_tile(vbase, 16 * t, NSEQ, j, lane);
      acc[t] = wmma_bf16(pt.v, bv, acc[t]);
    }
  }

#pragma unroll
  for (int v = 0; v < 8; ++v) Ls[v] = 1.0f / Ls[v];
#pragma unroll
  for (int t = 0; t < 4; ++t)
#pragma unroll
    for (int v = 0; v < 8; ++v)
      O[(size_t)(n0 + v + 8 * hi) * DMODEL + h * DHEAD + 16 * t + n] =
          (bf16)(acc[t][v] * Ls[v]);
}

// ------------------------------------------------------------------------------
extern "C" void kernel_launch(void* const* d_in, const int* in_sizes, int n_in,
                              void* d_out, int out_size, void* d_ws, size_t ws_size,
                              hipStream_t stream) {
  const float* x     = (const float*)d_in[0];
  const float* Wq    = (const float*)d_in[1];
  const float* bq    = (const float*)d_in[2];
  const float* Wk    = (const float*)d_in[3];
  const float* bk    = (const float*)d_in[4];
  const float* Wv    = (const float*)d_in[5];
  const float* bv    = (const float*)d_in[6];
  const float* Wo    = (const float*)d_in[7];
  const float* bo    = (const float*)d_in[8];
  const float* ln1_g = (const float*)d_in[9];
  const float* ln1_b = (const float*)d_in[10];
  const float* ln2_g = (const float*)d_in[11];
  const float* ln2_b = (const float*)d_in[12];
  const float* W1    = (const float*)d_in[13];
  const float* b1    = (const float*)d_in[14];
  const float* W2    = (const float*)d_in[15];
  const float* b2    = (const float*)d_in[16];

  char* ws = (char*)d_ws;
  size_t off = 0;
  auto alloc = [&](size_t bytes) -> char* {
    char* p = ws + off;
    off += (bytes + 255) & ~(size_t)255;
    return p;
  };
  const size_t DD = (size_t)DMODEL * DMODEL;   // 589824
  const size_t DF = (size_t)DMODEL * FDIM;     // 2359296
  const size_t ND = (size_t)NSEQ * DMODEL;     // 3145728
  const size_t NF = (size_t)NSEQ * FDIM;       // 12582912

  bf16* WqT  = (bf16*)alloc(DD * 2);
  bf16* WkT  = (bf16*)alloc(DD * 2);
  bf16* WvT  = (bf16*)alloc(DD * 2);
  bf16* WoT  = (bf16*)alloc(DD * 2);
  bf16* W1T  = (bf16*)alloc(DF * 2);
  bf16* W2T  = (bf16*)alloc(DF * 2);
  bf16* hbf  = (bf16*)alloc(ND * 2);   // LN1 out; reused for LN2 out
  bf16* qbf  = (bf16*)alloc(ND * 2);
  bf16* kbf  = (bf16*)alloc(ND * 2);
  bf16* vtbf = (bf16*)alloc(ND * 2);   // V transposed [768, 4096]
  bf16* abf  = (bf16*)alloc(ND * 2);   // attention output (concat heads)
  float* x1  = (float*)alloc(ND * 4);  // post-attention residual (f32)
  bf16* ff1  = (bf16*)alloc(NF * 2);   // gelu(h2@W1+b1)
  (void)ws_size; (void)in_sizes; (void)n_in; (void)out_size;

  // weight conversion
  conv_qkv_w<<<DD / 256, 256, 0, stream>>>(Wq, WqT);
  conv_qkv_w<<<DD / 256, 256, 0, stream>>>(Wk, WkT);
  conv_qkv_w<<<DD / 256, 256, 0, stream>>>(Wv, WvT);
  transpose_w<<<DD / 256, 256, 0, stream>>>(Wo, WoT, DMODEL, DMODEL);
  transpose_w<<<DF / 256, 256, 0, stream>>>(W1, W1T, DMODEL, FDIM);  // -> [F, D]
  transpose_w<<<DF / 256, 256, 0, stream>>>(W2, W2T, FDIM, DMODEL);  // -> [D, F]

  // LN1
  ln_kernel<<<NSEQ, 256, 0, stream>>>(x, ln1_g, ln1_b, hbf);

  // QKV projections (bias from bq/bk/bv flat [H*DH])
  dim3 gD(NSEQ / 128, DMODEL / 64);
  gemm_wmma<0><<<gD, 128, 0, stream>>>(hbf, WqT, bq, nullptr, qbf, NSEQ, DMODEL, DMODEL);
  gemm_wmma<0><<<gD, 128, 0, stream>>>(hbf, WkT, bk, nullptr, kbf, NSEQ, DMODEL, DMODEL);
  gemm_wmma<1><<<gD, 128, 0, stream>>>(hbf, WvT, bv, nullptr, vtbf, NSEQ, DMODEL, DMODEL);

  // flash attention
  attn_kernel<<<dim3(NSEQ / 64, NHEAD), 128, 0, stream>>>(qbf, kbf, vtbf, abf);

  // output projection + residual -> x1 (f32)
  gemm_wmma<2><<<gD, 128, 0, stream>>>(abf, WoT, bo, x, x1, NSEQ, DMODEL, DMODEL);

  // LN2 (reuse hbf)
  ln_kernel<<<NSEQ, 256, 0, stream>>>(x1, ln2_g, ln2_b, hbf);

  // FFN
  dim3 gF(NSEQ / 128, FDIM / 64);
  gemm_wmma<3><<<gF, 128, 0, stream>>>(hbf, W1T, b1, nullptr, ff1, NSEQ, DMODEL, FDIM);
  gemm_wmma<2><<<gD, 128, 0, stream>>>(ff1, W2T, b2, x1, (float*)d_out, NSEQ, FDIM, DMODEL);
}
